// GatedDeltaNet_32263794328222
// MI455X (gfx1250) — compile-verified, matrix operand order checked by reading, not run on hardware
//
#include <hip/hip_runtime.h>
#include <math.h>

// ---------------- problem constants ----------------
#define B_   2
#define T_   2048
#define HID_ 2048
#define NK_  8
#define NV_  16
#define HD_  128
#define KW_  4
#define KD_  (NK_*HD_)        // 1024
#define VD_  (NV_*HD_)        // 2048
#define CD_  (2*KD_+VD_)      // 4096
#define EXP_ (NV_/NK_)        // 2
#define MROWS (B_*T_)         // 4096

typedef __attribute__((ext_vector_type(16))) _Float16 v16h;
typedef __attribute__((ext_vector_type(8)))  _Float16 v8h;
typedef __attribute__((ext_vector_type(8)))  float    v8f;
typedef __attribute__((ext_vector_type(4)))  int      v4i;

// ---------------- CDNA5 async global->LDS copy (16B per lane) ----------------
// probed signature: (AS1 v4i* gaddr, AS3 v4i* lds, int offset, int cpol)
typedef __attribute__((address_space(1))) v4i* gptr128_t;
typedef __attribute__((address_space(3))) v4i* lptr128_t;

#if defined(__has_builtin)
#  if __has_builtin(__builtin_amdgcn_global_load_async_to_lds_b128)
#    define ASYNC_TIER_BUILTIN 1
#  endif
#endif

#ifdef ASYNC_TIER_BUILTIN
#define ASYNC_COPY_B128(gp, lp)                                              \
    __builtin_amdgcn_global_load_async_to_lds_b128(                          \
        (gptr128_t)(gp), (lptr128_t)(lp), 0, 0)
#else
#define ASYNC_COPY_B128(gp, lp)                                              \
    asm volatile("global_load_async_to_lds_b128 %0, %1, off"                 \
                 :: "v"((unsigned)(unsigned long long)(lp)), "v"(gp)         \
                 : "memory")
#endif

__device__ __forceinline__ void wait_async0() {
#if defined(__has_builtin) && __has_builtin(__builtin_amdgcn_s_wait_asynccnt)
    __builtin_amdgcn_s_wait_asynccnt(0);
#else
    asm volatile("s_wait_asynccnt 0x0" ::: "memory");
#endif
}

// =====================================================================
// f32 -> f16 conversion (one-shot, 8 elems / thread)
// =====================================================================
__global__ __launch_bounds__(256)
void gdn_cvt_f16(const float* __restrict__ src, _Float16* __restrict__ dst) {
    const size_t i = ((size_t)blockIdx.x * 256 + threadIdx.x) * 8;
    float4 a = *(const float4*)(src + i);
    float4 b = *(const float4*)(src + i + 4);
    v8h h;
    h[0] = (_Float16)a.x; h[1] = (_Float16)a.y; h[2] = (_Float16)a.z; h[3] = (_Float16)a.w;
    h[4] = (_Float16)b.x; h[5] = (_Float16)b.y; h[6] = (_Float16)b.z; h[7] = (_Float16)b.w;
    *(v8h*)(dst + i) = h;
}

// =====================================================================
// C = A (MxK, f16) * W^T (W NxK row-major, f16), f32 accumulate/output.
// 256 threads = 8 waves; 128x128 C tile; wave grid 4x2; 2x4 wmma accs.
// Double-buffered LDS stages filled by async global->LDS b128 copies.
// =====================================================================
#define GBM 128
#define GBN 128
#define GBK 64
#define LDT 72   // padded LDS row stride (halfs): 144B, 16B-aligned rows

__device__ __forceinline__ void gemm_stage_copy(
        const _Float16* __restrict__ Ag, const _Float16* __restrict__ Wg,
        _Float16* sa, _Float16* sb,
        int rowBlk, int colBlk, int K, int k0, int tid) {
#pragma unroll
    for (int u = 0; u < 4; ++u) {
        const int c   = tid + 256 * u;      // 0..1023 chunk id
        const int row = c >> 3;             // 0..127
        const int col = (c & 7) << 3;       // 0,8,..,56 (halfs)
        const _Float16* ga = Ag + (size_t)(rowBlk + row) * K + k0 + col;
        const _Float16* gb = Wg + (size_t)(colBlk + row) * K + k0 + col;
        _Float16* la = sa + row * LDT + col;
        _Float16* lb = sb + row * LDT + col;
        ASYNC_COPY_B128(ga, la);
        ASYNC_COPY_B128(gb, lb);
    }
}

__global__ __launch_bounds__(256)
void gdn_gemm_f16_nt(const _Float16* __restrict__ A, const _Float16* __restrict__ W,
                     float* __restrict__ C, int M, int N, int K) {
    __shared__ _Float16 smA[2][GBM * LDT];
    __shared__ _Float16 smB[2][GBN * LDT];

    const int tid  = threadIdx.x;
    const int lane = tid & 31;
    const int wave = tid >> 5;
    const int wm   = wave & 3;    // 0..3 -> 32-row slice of tile
    const int wn   = wave >> 2;   // 0..1 -> 64-col slice of tile
    const int rowBlk = blockIdx.y * GBM;
    const int colBlk = blockIdx.x * GBN;

    v8f acc[2][4];
#pragma unroll
    for (int i = 0; i < 2; ++i)
#pragma unroll
        for (int j = 0; j < 4; ++j)
#pragma unroll
            for (int r = 0; r < 8; ++r) acc[i][j][r] = 0.0f;

    const int nst = K / GBK;
    gemm_stage_copy(A, W, &smA[0][0], &smB[0][0], rowBlk, colBlk, K, 0, tid);

    for (int s = 0; s < nst; ++s) {
        wait_async0();               // our stage-s copies have landed in LDS
        __syncthreads();             // everyone's copies visible; prev reads done
        if (s + 1 < nst)
            gemm_stage_copy(A, W, &smA[(s + 1) & 1][0], &smB[(s + 1) & 1][0],
                            rowBlk, colBlk, K, (s + 1) * GBK, tid);

        const _Float16* sa = &smA[s & 1][0];
        const _Float16* sb = &smB[s & 1][0];
#pragma unroll
        for (int kk = 0; kk < GBK; kk += 32) {
            const int klane = kk + ((lane >> 4) << 4);  // lanes 16-31 take K+16
            v16h afrag[2];
#pragma unroll
            for (int i = 0; i < 2; ++i) {
                const _Float16* p = sa + (wm*32 + i*16 + (lane & 15)) * LDT + klane;
                v8h lo = *(const v8h*)(p);
                v8h hi = *(const v8h*)(p + 8);
                afrag[i] = __builtin_shufflevector(lo, hi,
                    0,1,2,3,4,5,6,7,8,9,10,11,12,13,14,15);
            }
            v16h bfrag[4];
#pragma unroll
            for (int j = 0; j < 4; ++j) {
                const _Float16* p = sb + (wn*64 + j*16 + (lane & 15)) * LDT + klane;
                v8h lo = *(const v8h*)(p);
                v8h hi = *(const v8h*)(p + 8);
                bfrag[j] = __builtin_shufflevector(lo, hi,
                    0,1,2,3,4,5,6,7,8,9,10,11,12,13,14,15);
            }
#pragma unroll
            for (int i = 0; i < 2; ++i)
#pragma unroll
                for (int j = 0; j < 4; ++j)
                    acc[i][j] = __builtin_amdgcn_wmma_f32_16x16x32_f16(
                        false, afrag[i], false, bfrag[j],
                        (short)0, acc[i][j], false, false);
        }
    }

    // C/D layout: VGPR r, lanes 0-15 -> M=r, N=lane; lanes 16-31 -> M=r+8
#pragma unroll
    for (int i = 0; i < 2; ++i)
#pragma unroll
        for (int j = 0; j < 4; ++j) {
            const int col   = colBlk + wn*64 + j*16 + (lane & 15);
            const int rbase = rowBlk + wm*32 + i*16 + ((lane >> 4) << 3);
#pragma unroll
            for (int r = 0; r < 8; ++r)
                C[(size_t)(rbase + r) * N + col] = acc[i][j][r];
        }
}

// =====================================================================
// beta / decay projections (tiny GEMV) + activations
// =====================================================================
__global__ __launch_bounds__(32)
void gdn_bd_kernel(const float* __restrict__ x, const float* __restrict__ Wb,
                   const float* __restrict__ Wa, const float* __restrict__ dt_bias,
                   const float* __restrict__ A_log,
                   float* __restrict__ beta, float* __restrict__ decay) {
    const int m = blockIdx.x;
    const int j = threadIdx.x;
    const float* xr = x + (size_t)m * HID_;
    const float* w  = (j < NV_) ? (Wb + (size_t)j * HID_)
                                : (Wa + (size_t)(j - NV_) * HID_);
    float acc = 0.f;
    for (int i = 0; i < HID_; i += 4) {
        acc += xr[i+0]*w[i+0] + xr[i+1]*w[i+1] + xr[i+2]*w[i+2] + xr[i+3]*w[i+3];
    }
    if (j < NV_) {
        beta[(size_t)m * NV_ + j] = 1.f / (1.f + __expf(-acc));
    } else {
        const int h = j - NV_;
        float sp = acc + dt_bias[h];
        sp = (sp > 20.f) ? sp : log1pf(__expf(sp));
        decay[(size_t)m * NV_ + h] = __expf(-__expf(A_log[h]) * sp);
    }
}

// =====================================================================
// depthwise causal conv1d (K=4, left zero pad 3) + silu
// =====================================================================
__global__ __launch_bounds__(256)
void gdn_conv_silu(const float* __restrict__ qkv_raw, const float* __restrict__ conv_w,
                   float* __restrict__ qkv_act) {
    const size_t idx = (size_t)blockIdx.x * blockDim.x + threadIdx.x;
    const int c = (int)(idx % CD_);
    const int m = (int)(idx / CD_);
    const int t = m % T_;
    const int b = m / T_;
    float acc = 0.f;
#pragma unroll
    for (int jj = 0; jj < KW_; ++jj) {
        const int ts = t - (KW_ - 1) + jj;
        if (ts >= 0)
            acc += qkv_raw[((size_t)b * T_ + ts) * CD_ + c] * conv_w[c * KW_ + jj];
    }
    qkv_act[idx] = acc / (1.f + __expf(-acc));   // silu
}

// =====================================================================
// L2-normalize q (scaled by HD^-1/2) and k ; one wave per 128-vector
// =====================================================================
__global__ __launch_bounds__(256)
void gdn_qknorm(const float* __restrict__ qkv_act,
                float* __restrict__ qn, float* __restrict__ kn) {
    const int lane = threadIdx.x & 31;
    const int wv   = blockIdx.x * (blockDim.x >> 5) + (threadIdx.x >> 5);
    const int m    = wv >> 4;      // / (2*NK_) = 16
    const int r    = wv & 15;
    const int isK  = r >> 3;
    const int head = r & 7;
    const float* src = qkv_act + (size_t)m * CD_ + (size_t)isK * KD_ + head * HD_;
    float v0[4]; float ss = 0.f;
#pragma unroll
    for (int u = 0; u < 4; ++u) { v0[u] = src[lane + 32*u]; ss += v0[u]*v0[u]; }
#pragma unroll
    for (int off = 16; off > 0; off >>= 1) ss += __shfl_xor(ss, off, 32);
    const float denom = fmaxf(sqrtf(ss), 1e-12f);
    const float scale = (isK ? 1.0f : 0.08838834764831845f) / denom; // HD^-0.5
    float* dst = (isK ? kn : qn) + ((size_t)m * NK_ + head) * HD_;
#pragma unroll
    for (int u = 0; u < 4; ++u) dst[lane + 32*u] = v0[u] * scale;
}

// =====================================================================
// gated delta-rule recurrence. One block = one (b, head) instance.
// 128 threads; thread j owns state column S[:, j] in 128 VGPRs.
// =====================================================================
__global__ __launch_bounds__(128)
void gdn_scan(const float* __restrict__ qn, const float* __restrict__ kn,
              const float* __restrict__ qkv_act,
              const float* __restrict__ beta, const float* __restrict__ decay,
              float* __restrict__ attn) {
    const int b  = blockIdx.x / NV_;
    const int h  = blockIdx.x % NV_;
    const int kh = h / EXP_;
    const int j  = threadIdx.x;      // 0..127 : state column

    __shared__ float sK[HD_];
    __shared__ float sQ[HD_];

    float S[HD_];
#pragma unroll
    for (int i = 0; i < HD_; ++i) S[i] = 0.f;

    for (int t = 0; t < T_; ++t) {
        const size_t m = (size_t)b * T_ + t;
        const float kv  = kn[(m * NK_ + kh) * HD_ + j];
        const float qv  = qn[(m * NK_ + kh) * HD_ + j];
        const float vv  = qkv_act[m * CD_ + 2*KD_ + (size_t)h * HD_ + j];
        const float bet = beta [m * NV_ + h];
        const float dec = decay[m * NV_ + h];

        __syncthreads();
        sK[j] = kv; sQ[j] = qv;
        __syncthreads();

        float sk = 0.f;
#pragma unroll
        for (int i = 0; i < HD_; ++i) sk += sK[i] * S[i];

        const float delta = bet * (vv - dec * sk);

        float o = 0.f;
#pragma unroll
        for (int i = 0; i < HD_; ++i) {
            const float s = S[i] * dec + sK[i] * delta;
            S[i] = s;
            o += sQ[i] * s;
        }
        attn[(m * NV_ + h) * HD_ + j] = o;
    }
}

// =====================================================================
// gated RMSNorm over head dim:  y16 = f16( rmsnorm(attn)*w*silu(z) )
// =====================================================================
__global__ __launch_bounds__(256)
void gdn_rmsgate(const float* __restrict__ attn, const float* __restrict__ z,
                 const float* __restrict__ norm_w, _Float16* __restrict__ y16) {
    const int lane = threadIdx.x & 31;
    const int row  = blockIdx.x * (blockDim.x >> 5) + (threadIdx.x >> 5);
    const int m = row / NV_;
    const int h = row % NV_;
    const float* p = attn + (size_t)row * HD_;
    const float* zp = z + (size_t)m * VD_ + (size_t)h * HD_;
    _Float16* yp = y16 + (size_t)row * HD_;
    float v0[4]; float ss = 0.f;
#pragma unroll
    for (int u = 0; u < 4; ++u) { v0[u] = p[lane + 32*u]; ss += v0[u]*v0[u]; }
#pragma unroll
    for (int off = 16; off > 0; off >>= 1) ss += __shfl_xor(ss, off, 32);
    const float inv_rms = rsqrtf(ss * (1.0f / HD_) + 1e-6f);
#pragma unroll
    for (int u = 0; u < 4; ++u) {
        const int c = lane + 32*u;
        const float zz = zp[c];
        const float sig = zz / (1.f + __expf(-zz));
        yp[c] = (_Float16)(v0[u] * inv_rms * norm_w[c] * sig);
    }
}

// =====================================================================
extern "C" void kernel_launch(void* const* d_in, const int* in_sizes, int n_in,
                              void* d_out, int out_size, void* d_ws, size_t ws_size,
                              hipStream_t stream) {
    const float* x       = (const float*)d_in[0];
    const float* W_qkv   = (const float*)d_in[1];
    const float* W_z     = (const float*)d_in[2];
    const float* W_b     = (const float*)d_in[3];
    const float* W_a     = (const float*)d_in[4];
    const float* conv_w  = (const float*)d_in[5];
    const float* dt_bias = (const float*)d_in[6];
    const float* A_log   = (const float*)d_in[7];
    const float* norm_w  = (const float*)d_in[8];
    const float* W_out   = (const float*)d_in[9];
    float* out = (float*)d_out;

    float* ws = (float*)d_ws;
    size_t o = 0;
    float* qkv_raw = ws + o; o += (size_t)MROWS * CD_;        // f32
    float* qkv_act = ws + o; o += (size_t)MROWS * CD_;
    float* zbuf    = ws + o; o += (size_t)MROWS * VD_;
    float* beta    = ws + o; o += (size_t)MROWS * NV_;
    float* decay   = ws + o; o += (size_t)MROWS * NV_;
    float* qnb     = ws + o; o += (size_t)MROWS * NK_ * HD_;
    float* knb     = ws + o; o += (size_t)MROWS * NK_ * HD_;
    float* attn    = ws + o; o += (size_t)MROWS * NV_ * HD_;

    _Float16* hb = (_Float16*)(ws + o);
    size_t ho = 0;
    _Float16* x16    = hb + ho; ho += (size_t)MROWS * HID_;   // 8.4M halfs
    _Float16* wqkv16 = hb + ho; ho += (size_t)CD_ * HID_;     // 8.4M
    _Float16* wz16   = hb + ho; ho += (size_t)VD_ * HID_;     // 4.2M
    _Float16* wout16 = hb + ho; ho += (size_t)HID_ * VD_;     // 4.2M
    _Float16* y16    = hb + ho; ho += (size_t)MROWS * VD_;    // 8.4M

    // 0) one-shot f16 conversions for WMMA operands
    gdn_cvt_f16<<<(int)(((size_t)MROWS * HID_) / 2048), 256, 0, stream>>>(x, x16);
    gdn_cvt_f16<<<(int)(((size_t)CD_ * HID_) / 2048), 256, 0, stream>>>(W_qkv, wqkv16);
    gdn_cvt_f16<<<(int)(((size_t)VD_ * HID_) / 2048), 256, 0, stream>>>(W_z, wz16);
    gdn_cvt_f16<<<(int)(((size_t)HID_ * VD_) / 2048), 256, 0, stream>>>(W_out, wout16);

    // 1) qkv = x @ W_qkv^T    (4096 x 4096, K=2048)
    gdn_gemm_f16_nt<<<dim3(CD_/GBN, MROWS/GBM), 256, 0, stream>>>(
        x16, wqkv16, qkv_raw, MROWS, CD_, HID_);
    // 2) z = x @ W_z^T        (4096 x 2048, K=2048)
    gdn_gemm_f16_nt<<<dim3(VD_/GBN, MROWS/GBM), 256, 0, stream>>>(
        x16, wz16, zbuf, MROWS, VD_, HID_);
    // 3) beta / decay
    gdn_bd_kernel<<<MROWS, 32, 0, stream>>>(x, W_b, W_a, dt_bias, A_log, beta, decay);
    // 4) causal depthwise conv + silu
    gdn_conv_silu<<<(int)(((size_t)MROWS * CD_) / 256), 256, 0, stream>>>(
        qkv_raw, conv_w, qkv_act);
    // 5) q/k L2 norm
    gdn_qknorm<<<(MROWS * 2 * NK_) / 8, 256, 0, stream>>>(qkv_act, qnb, knb);
    // 6) sequential gated delta-rule scan
    gdn_scan<<<B_ * NV_, 128, 0, stream>>>(qnb, knb, qkv_act, beta, decay, attn);
    // 7) gated RMSNorm -> f16
    gdn_rmsgate<<<(MROWS * NV_) / 8, 256, 0, stream>>>(attn, zbuf, norm_w, y16);
    // 8) out = y @ W_out^T  (4096 x 2048, K=2048)
    gdn_gemm_f16_nt<<<dim3(HID_/GBN, MROWS/GBM), 256, 0, stream>>>(
        y16, wout16, out, MROWS, HID_, VD_);
}
